// GAT25Model_6124623364733
// MI455X (gfx1250) — compile-verified, matrix operand order checked by reading, not run on hardware
//
#include <hip/hip_runtime.h>
#include <math.h>

typedef __attribute__((ext_vector_type(16))) __bf16 v16bf;
typedef __attribute__((ext_vector_type(8)))  __bf16 v8bf;
typedef __attribute__((ext_vector_type(8)))  float  v8f;

#define NEG_SLOPE 0.2f
#define BN_EPS 1e-5f

__device__ __forceinline__ __bf16 f2bf(float x) {
    unsigned u = __float_as_uint(x);
    u += 0x7FFFu + ((u >> 16) & 1u);        // round-to-nearest-even
    unsigned short h = (unsigned short)(u >> 16);
    __bf16 r;
    __builtin_memcpy(&r, &h, 2);
    return r;
}

__device__ __forceinline__ void atomicMaxF(float* addr, float val) {
    if (val >= 0.0f) atomicMax((int*)addr, __float_as_int(val));
    else             atomicMin((unsigned int*)addr, __float_as_uint(val));
}

// ---------------- generic fill ----------------
__global__ void k_fill(float* p, size_t n, float v) {
    size_t i = (size_t)blockIdx.x * 256 + threadIdx.x;
    if (i < n) p[i] = v;
}

// ------- pack W1l/W1r (f32 row-major 128x64) into WMMA-B fragment order -------
// layout: P[w][kt][n][h][i], i=0..15, element k = kt*32 + (i/8)*16 + h*8 + (i&7)
__global__ void k_packW(const float* __restrict__ W1l, const float* __restrict__ W1r,
                        __bf16* __restrict__ Wp) {
    int idx = blockIdx.x * 256 + threadIdx.x;
    if (idx >= 2 * 4 * 64 * 2 * 16) return;
    int i  = idx & 15;
    int h  = (idx >> 4) & 1;
    int n  = (idx >> 5) & 63;
    int kt = (idx >> 11) & 3;
    int w  = idx >> 13;
    int k  = kt * 32 + (i >> 3) * 16 + h * 8 + (i & 7);
    const float* W = w ? W1r : W1l;
    Wp[idx] = f2bf(W[k * 64 + n]);
}

// ------- layer-1 dual GEMM: xl = X@W1l+b1l, xr = X@W1r+b1r via bf16 WMMA -------
// block = 256 threads = 8 waves; one 16-row tile of X; wave -> {Wl,Wr} x 4 col tiles
__global__ __launch_bounds__(256) void k_gemm1(
    const float* __restrict__ X, const __bf16* __restrict__ Wp,
    const float* __restrict__ b1l, const float* __restrict__ b1r,
    float* __restrict__ xl1, float* __restrict__ xr1, int N)
{
    __shared__ __align__(16) __bf16 sX[16][136];   // padded row stride (272B, 16B-aligned)
    const int row0 = blockIdx.x * 16;
    const int t = threadIdx.x;

    // cooperative load of 16x128 f32 tile -> bf16 LDS (8 elems/thread)
    {
        int r  = t >> 4;
        int c0 = (t & 15) * 8;
        int rr = row0 + r; if (rr > N - 1) rr = N - 1;   // clamp (branch-free, keeps EXEC full)
        const float* xp = X + (size_t)rr * 128 + c0;
#pragma unroll
        for (int i = 0; i < 8; ++i) sX[r][c0 + i] = f2bf(xp[i]);
    }
    __syncthreads();

    const int wave = t >> 5;          // wave32
    const int lane = t & 31;
    const int wsel = wave >> 2;       // 0 -> Wl, 1 -> Wr
    const int n0   = (wave & 3) * 16;
    const int m    = lane & 15;       // A row; also B/D column-in-tile
    const int h    = lane >> 4;

    v8f acc = {};
#pragma unroll
    for (int kt = 0; kt < 4; ++kt) {
        const int k0 = kt * 32;
        // A fragment: lane holds row m, k = k0 + {h*8..h*8+7, 16+h*8..16+h*8+7}
        v8bf lo = *(const v8bf*)&sX[m][k0 + h * 8];
        v8bf hi = *(const v8bf*)&sX[m][k0 + 16 + h * 8];
        v16bf a;
#pragma unroll
        for (int i = 0; i < 8; ++i) { a[i] = lo[i]; a[i + 8] = hi[i]; }
        // B fragment: one contiguous 32B load from packed weights
        const __bf16* bp = Wp + ((((size_t)(wsel * 4 + kt) * 64 + (n0 + m)) * 2 + h) << 4);
        v16bf b = *(const v16bf*)bp;
        acc = __builtin_amdgcn_wmma_f32_16x16x32_bf16(false, a, false, b,
                                                      (short)0, acc, false, false);
    }

    const float bv = (wsel ? b1r : b1l)[n0 + m];
    float* out = wsel ? xr1 : xl1;
    const int n = n0 + m;
    if (row0 + 16 <= N) {
        // fast path: uniform full tile, straight-line coalesced stores
#pragma unroll
        for (int r = 0; r < 8; ++r) {
            int mm = r + 8 * h;                    // C/D layout: vgpr r, lane-half h
            out[(size_t)(row0 + mm) * 64 + n] = acc[r] + bv;
        }
    } else {
#pragma unroll
        for (int r = 0; r < 8; ++r) {
            int mm = r + 8 * h;
            if (row0 + mm < N) out[(size_t)(row0 + mm) * 64 + n] = acc[r] + bv;
        }
    }
}

// ---------------- layer-1 edge passes ----------------
__global__ void k_edge_logit1(const int* __restrict__ src, const int* __restrict__ dst,
                              const float* __restrict__ xl, const float* __restrict__ xr,
                              const float* __restrict__ att,
                              float* __restrict__ elog, float* __restrict__ smax, int E)
{
    int e = blockIdx.x * 256 + threadIdx.x;
    if (e >= E) return;
    int s = src[e], d = dst[e];
    // node rows are 256B-aligned (base + k*64 floats): force b128 loads
    const float4* ps = (const float4*)(xl + (size_t)s * 64);
    const float4* pd = (const float4*)(xr + (size_t)d * 64);
    const float4* pa = (const float4*)att;
    float acc = 0.f;
#pragma unroll
    for (int c4 = 0; c4 < 16; ++c4) {
        float4 a = ps[c4], b = pd[c4], w = pa[c4];
        float v0 = a.x + b.x, v1 = a.y + b.y, v2 = a.z + b.z, v3 = a.w + b.w;
        v0 = v0 > 0.f ? v0 : NEG_SLOPE * v0;
        v1 = v1 > 0.f ? v1 : NEG_SLOPE * v1;
        v2 = v2 > 0.f ? v2 : NEG_SLOPE * v2;
        v3 = v3 > 0.f ? v3 : NEG_SLOPE * v3;
        acc += v0 * w.x + v1 * w.y + v2 * w.z + v3 * w.w;
    }
    elog[e] = acc;
    atomicMaxF(smax + d, acc);
}

__global__ void k_edge_exp(const int* __restrict__ dst, const float* __restrict__ elog,
                           const float* __restrict__ smax, float* __restrict__ ealp,
                           float* __restrict__ ssum, int E)
{
    int e = blockIdx.x * 256 + threadIdx.x;
    if (e >= E) return;
    int d = dst[e];
    float a = expf(elog[e] - smax[d]);
    ealp[e] = a;
    atomicAdd(ssum + d, a);
}

__global__ void k_edge_scatter1(const int* __restrict__ src, const int* __restrict__ dst,
                                const float* __restrict__ xl, const float* __restrict__ ealp,
                                const float* __restrict__ ssum, float* __restrict__ x1o, int E)
{
    long long idx = (long long)blockIdx.x * 256 + threadIdx.x;
    if (idx >= (long long)E * 64) return;
    int e = (int)(idx >> 6);
    int c = (int)(idx & 63);
    int s = src[e], d = dst[e];
    float alpha = ealp[e] / (ssum[d] + 1e-16f);
    atomicAdd(&x1o[(size_t)d * 64 + c], alpha * xl[(size_t)s * 64 + c]);
}

// ---------------- batch norm ----------------
__global__ void k_bnstats(const float* __restrict__ x1, const float* __restrict__ bias1,
                          float* __restrict__ bnsum, float* __restrict__ bnsq, int N)
{
    __shared__ float sS[256], sQ[256];
    int c   = threadIdx.x & 63;
    int rep = threadIdx.x >> 6;           // 4 replicas per channel
    float b = bias1[c];
    float s = 0.f, q = 0.f;
    for (int i = blockIdx.x * 4 + rep; i < N; i += gridDim.x * 4) {
        float v = x1[(size_t)i * 64 + c] + b;
        s += v; q += v * v;
    }
    sS[threadIdx.x] = s; sQ[threadIdx.x] = q;
    __syncthreads();
    if (rep == 0) {
        s = sS[c] + sS[c + 64] + sS[c + 128] + sS[c + 192];
        q = sQ[c] + sQ[c + 64] + sQ[c + 128] + sQ[c + 192];
        atomicAdd(bnsum + c, s);
        atomicAdd(bnsq + c, q);
    }
}

__global__ void k_bnapply(float* __restrict__ x1, const float* __restrict__ bias1,
                          const float* __restrict__ bnsum, const float* __restrict__ bnsq,
                          const float* __restrict__ gamma, const float* __restrict__ beta, int N)
{
    long long idx = (long long)blockIdx.x * 256 + threadIdx.x;
    if (idx >= (long long)N * 64) return;
    int c = (int)(idx & 63);
    float mean = bnsum[c] / (float)N;
    float var  = bnsq[c] / (float)N - mean * mean;
    float v = x1[idx] + bias1[c];
    x1[idx] = (v - mean) * rsqrtf(var + BN_EPS) * gamma[c] + beta[c];
}

// ---------------- layer-2 node transform (C=64 -> 1) ----------------
__global__ void k_lin2(const float* __restrict__ x1,
                       const float* __restrict__ W2l, const float* __restrict__ b2l,
                       const float* __restrict__ W2r, const float* __restrict__ b2r,
                       float* __restrict__ x2l, float* __restrict__ x2r, int N)
{
    int i = blockIdx.x * 256 + threadIdx.x;
    if (i >= N) return;
    const float4* p  = (const float4*)(x1 + (size_t)i * 64);
    const float4* wl = (const float4*)W2l;
    const float4* wr = (const float4*)W2r;
    float al = 0.f, ar = 0.f;
#pragma unroll
    for (int c4 = 0; c4 < 16; ++c4) {
        float4 v = p[c4], a = wl[c4], b = wr[c4];
        al += v.x * a.x + v.y * a.y + v.z * a.z + v.w * a.w;
        ar += v.x * b.x + v.y * b.y + v.z * b.z + v.w * b.w;
    }
    x2l[i] = al + b2l[0];
    x2r[i] = ar + b2r[0];
}

// ---------------- layer-2 edge passes (scalar channel) ----------------
__global__ void k_edge_logit2(const int* __restrict__ src, const int* __restrict__ dst,
                              const float* __restrict__ x2l, const float* __restrict__ x2r,
                              const float* __restrict__ att2,
                              float* __restrict__ elog, float* __restrict__ smax, int E)
{
    int e = blockIdx.x * 256 + threadIdx.x;
    if (e >= E) return;
    int s = src[e], d = dst[e];
    float v = x2l[s] + x2r[d];
    v = v > 0.f ? v : NEG_SLOPE * v;
    float acc = v * att2[0];
    elog[e] = acc;
    atomicMaxF(smax + d, acc);
}

__global__ void k_edge_scatter2(const int* __restrict__ src, const int* __restrict__ dst,
                                const float* __restrict__ x2l, const float* __restrict__ ealp,
                                const float* __restrict__ ssum, float* __restrict__ out2, int E)
{
    int e = blockIdx.x * 256 + threadIdx.x;
    if (e >= E) return;
    int s = src[e], d = dst[e];
    float alpha = ealp[e] / (ssum[d] + 1e-16f);
    atomicAdd(&out2[d], alpha * x2l[s]);
}

// ---------------- final mean ----------------
__global__ void k_mean(const float* __restrict__ out2, const float* __restrict__ bias2,
                       float* __restrict__ dout, int N)
{
    __shared__ float sred[256];
    int i = blockIdx.x * 256 + threadIdx.x;
    float v = 0.f;
    if (i < N) v = out2[i] + bias2[0];
    sred[threadIdx.x] = v;
    __syncthreads();
    for (int s = 128; s > 0; s >>= 1) {
        if (threadIdx.x < s) sred[threadIdx.x] += sred[threadIdx.x + s];
        __syncthreads();
    }
    if (threadIdx.x == 0) atomicAdd(dout, sred[0] / (float)N);
}

extern "C" void kernel_launch(void* const* d_in, const int* in_sizes, int n_in,
                              void* d_out, int out_size, void* d_ws, size_t ws_size,
                              hipStream_t stream)
{
    const float* X    = (const float*)d_in[0];
    const int*   ei   = (const int*)d_in[1];
    const float* W1l  = (const float*)d_in[2];
    const float* b1l  = (const float*)d_in[3];
    const float* W1r  = (const float*)d_in[4];
    const float* b1r  = (const float*)d_in[5];
    const float* att1 = (const float*)d_in[6];
    const float* bias1= (const float*)d_in[7];
    const float* gamma= (const float*)d_in[8];
    const float* beta = (const float*)d_in[9];
    const float* W2l  = (const float*)d_in[10];
    const float* b2l  = (const float*)d_in[11];
    const float* W2r  = (const float*)d_in[12];
    const float* b2r  = (const float*)d_in[13];
    const float* att2 = (const float*)d_in[14];
    const float* bias2= (const float*)d_in[15];

    const int N = in_sizes[0] / 128;
    const int E = in_sizes[1] / 2;
    const int* srcI = ei;
    const int* dstI = ei + E;

    float* ws = (float*)d_ws;
    size_t off = 0;
    float* xl1  = ws + off; off += (size_t)N * 64;
    float* xr1  = ws + off; off += (size_t)N * 64;
    float* x1o  = ws + off; off += (size_t)N * 64;
    float* elog = ws + off; off += (size_t)E;
    float* ealp = ws + off; off += (size_t)E;
    float* smax = ws + off; off += (size_t)N;
    float* ssum = ws + off; off += (size_t)N;
    float* x2l  = ws + off; off += (size_t)N;
    float* x2r  = ws + off; off += (size_t)N;
    float* out2 = ws + off; off += (size_t)N;
    float* bnsum= ws + off; off += 64;
    float* bnsq = ws + off; off += 64;
    off = (off + 7) & ~(size_t)7;                 // 32B-align packed weights
    __bf16* wpk = (__bf16*)(ws + off);

    const int B = 256;
    dim3 blk(B);
    const int gN    = (N + B - 1) / B;
    const int gE    = (E + B - 1) / B;
    const int gN64  = (int)(((long long)N * 64 + B - 1) / B);
    const int gE64  = (int)(((long long)E * 64 + B - 1) / B);
    const int gTile = (N + 15) / 16;

    // ---- init scratch / output (deterministic each call) ----
    k_fill<<<gN64, blk, 0, stream>>>(x1o, (size_t)N * 64, 0.f);
    k_fill<<<gN, blk, 0, stream>>>(smax, (size_t)N, -INFINITY);
    k_fill<<<gN, blk, 0, stream>>>(ssum, (size_t)N, 0.f);
    k_fill<<<gN, blk, 0, stream>>>(out2, (size_t)N, 0.f);
    k_fill<<<1, blk, 0, stream>>>(bnsum, 64, 0.f);
    k_fill<<<1, blk, 0, stream>>>(bnsq, 64, 0.f);
    k_fill<<<1, blk, 0, stream>>>((float*)d_out, 1, 0.f);

    // ---- layer 1: dense transforms via WMMA ----
    k_packW<<<64, blk, 0, stream>>>(W1l, W1r, wpk);
    k_gemm1<<<gTile, blk, 0, stream>>>(X, wpk, b1l, b1r, xl1, xr1, N);

    // ---- layer 1: edge attention (L2-resident gathers) ----
    k_edge_logit1<<<gE, blk, 0, stream>>>(srcI, dstI, xl1, xr1, att1, elog, smax, E);
    k_edge_exp<<<gE, blk, 0, stream>>>(dstI, elog, smax, ealp, ssum, E);
    k_edge_scatter1<<<gE64, blk, 0, stream>>>(srcI, dstI, xl1, ealp, ssum, x1o, E);

    // ---- batch norm ----
    k_bnstats<<<1024, blk, 0, stream>>>(x1o, bias1, bnsum, bnsq, N);
    k_bnapply<<<gN64, blk, 0, stream>>>(x1o, bias1, bnsum, bnsq, gamma, beta, N);

    // ---- layer 2 ----
    k_lin2<<<gN, blk, 0, stream>>>(x1o, W2l, b2l, W2r, b2r, x2l, x2r, N);
    k_fill<<<gN, blk, 0, stream>>>(smax, (size_t)N, -INFINITY);
    k_fill<<<gN, blk, 0, stream>>>(ssum, (size_t)N, 0.f);
    k_edge_logit2<<<gE, blk, 0, stream>>>(srcI, dstI, x2l, x2r, att2, elog, smax, E);
    k_edge_exp<<<gE, blk, 0, stream>>>(dstI, elog, smax, ealp, ssum, E);
    k_edge_scatter2<<<gE, blk, 0, stream>>>(srcI, dstI, x2l, ealp, ssum, out2, E);

    // ---- final mean ----
    k_mean<<<gN, blk, 0, stream>>>(out2, bias2, (float*)d_out, N);
}
